// PatternSeparator_58737972740313
// MI455X (gfx1250) — matrix operand order repaired and checked by reading, not compiled, not run on hardware
//
#include <hip/hip_runtime.h>
#include <math.h>

#define BATCH   8192
#define IN_DIM  512
#define RFF     4096
#define TOPK    409
#define RFF_SCALE 0.022097086912079608f  /* sqrt(2/4096) */
#define NCHUNK  (IN_DIM / 32)

typedef __attribute__((ext_vector_type(16))) __bf16 v16bf;
typedef __attribute__((ext_vector_type(8)))  __bf16 v8bf;
typedef __attribute__((ext_vector_type(8)))  float  v8f;
typedef __attribute__((ext_vector_type(4)))  int    v4i;

#define GAS __attribute__((address_space(1)))
#define LAS __attribute__((address_space(3)))

// ---- CDNA5 async global->LDS copy (ASYNCcnt-tracked), 16 bytes -------------
#if __has_builtin(__builtin_amdgcn_global_load_async_to_lds_b128)
__device__ __forceinline__ void async_cp16(void* lds, const void* glob) {
  __builtin_amdgcn_global_load_async_to_lds_b128(
      (GAS v4i*)glob, (LAS v4i*)lds, 0, 0);
}
#else
__device__ __forceinline__ void async_cp16(void* lds, const void* glob) {
  unsigned l = (unsigned)(unsigned long long)(LAS void*)lds;
  unsigned long long g = (unsigned long long)glob;
  asm volatile("global_load_async_to_lds_b128 %0, %1, off"
               :: "v"(l), "v"(g) : "memory");
}
#endif

#if __has_builtin(__builtin_amdgcn_s_wait_asynccnt)
#define WAIT_ASYNC() __builtin_amdgcn_s_wait_asynccnt(0)
#else
#define WAIT_ASYNC() asm volatile("s_wait_asynccnt 0x0" ::: "memory")
#endif

// ---------------------------------------------------------------------------
// Pre-pass: split f32 -> bf16 hi + bf16 residual lo (bf16x3 emulation inputs)
// ---------------------------------------------------------------------------
__global__ __launch_bounds__(256) void split_f32_bf16(
    const float* __restrict__ src, __bf16* __restrict__ hi,
    __bf16* __restrict__ lo, int n)
{
  int i = (blockIdx.x * 256 + threadIdx.x) * 4;
  if (i + 3 < n) {
    float4 f = *(const float4*)(src + i);
#pragma unroll
    for (int e = 0; e < 4; ++e) {
      float v = (&f.x)[e];
      __bf16 h = (__bf16)v;
      hi[i + e] = h;
      lo[i + e] = (__bf16)(v - (float)h);
    }
  }
}

// ---------------------------------------------------------------------------
// Fast GEMM: pre-split bf16 operands, async-LDS double-buffered B panel.
// Block 256 thr = 8 waves; block tile 128(M) x 64(N); wave tile 32x32.
// ---------------------------------------------------------------------------
__global__ __launch_bounds__(256) void rff_gemm_cos_bf16(
    const __bf16* __restrict__ xh, const __bf16* __restrict__ xl,
    const __bf16* __restrict__ wh, const __bf16* __restrict__ wl,
    const float* __restrict__ bias, float* __restrict__ out)
{
  // [buffer][hi/lo][col*32 + k] for one 64-col x 32-K chunk of B = W^T
  __shared__ __bf16 ldsB[2][2][64 * 32];

  const int tid    = threadIdx.x;
  const int lane   = tid & 31;
  const int wave   = tid >> 5;
  const int wm     = wave & 3;
  const int wn     = wave >> 2;
  const int l15    = lane & 15;
  const bool hiHalf = lane >= 16;

  const int rowBase = blockIdx.x * 128 + wm * 32;
  const int colBlk  = blockIdx.y * 64;          // B panel base col
  const int colBase = colBlk + wn * 32;

  // B-panel staging assignment: thread -> (col 0..63, k-seg of 8 bf16)
  const int sc = tid >> 2;
  const int sk = (tid & 3) * 8;
  const __bf16* gwh = wh + (size_t)(colBlk + sc) * IN_DIM + sk;
  const __bf16* gwl = wl + (size_t)(colBlk + sc) * IN_DIM + sk;

  v8f acc[2][2];
#pragma unroll
  for (int i = 0; i < 2; ++i)
#pragma unroll
    for (int j = 0; j < 2; ++j)
      acc[i][j] = (v8f){0.f, 0.f, 0.f, 0.f, 0.f, 0.f, 0.f, 0.f};

  // A-fragment row pointers (lane half selects K-contiguous runs
  // {0..15} vs {16..31}, matching the B fragment order)
  const int aOff = hiHalf ? 16 : 0;
  const __bf16* xrh[2];
  const __bf16* xrl[2];
#pragma unroll
  for (int i = 0; i < 2; ++i) {
    size_t r = (size_t)(rowBase + i * 16 + l15) * IN_DIM + aOff;
    xrh[i] = xh + r;
    xrl[i] = xl + r;
  }

  // issue chunk 0 staging
  async_cp16(&ldsB[0][0][sc * 32 + sk], gwh);
  async_cp16(&ldsB[0][1][sc * 32 + sk], gwl);

  for (int c = 0; c < NCHUNK; ++c) {
    const int kc = c * 32;
    const int buf = c & 1;

    WAIT_ASYNC();       // this wave's staging for buffer `buf` landed
    __syncthreads();    // all waves' staging landed; prior reads of other buf done

    if (c + 1 < NCHUNK) {
      async_cp16(&ldsB[buf ^ 1][0][sc * 32 + sk], gwh + kc + 32);
      async_cp16(&ldsB[buf ^ 1][1][sc * 32 + sk], gwl + kc + 32);
    }

    // A fragments: two contiguous 16B runs per lane, direct from global
    v16bf aHi[2], aLo[2];
#pragma unroll
    for (int i = 0; i < 2; ++i) {
      v8bf h0 = *(const v8bf*)(xrh[i] + kc);
      v8bf h1 = *(const v8bf*)(xrh[i] + kc + 8);
      v8bf l0 = *(const v8bf*)(xrl[i] + kc);
      v8bf l1 = *(const v8bf*)(xrl[i] + kc + 8);
      aHi[i] = __builtin_shufflevector(h0, h1, 0,1,2,3,4,5,6,7,8,9,10,11,12,13,14,15);
      aLo[i] = __builtin_shufflevector(l0, l1, 0,1,2,3,4,5,6,7,8,9,10,11,12,13,14,15);
    }

    // B fragments from LDS: K-contiguous 32B per lane (matches B layout)
    v16bf bHi[2], bLo[2];
#pragma unroll
    for (int j = 0; j < 2; ++j) {
      const int colLocal = wn * 32 + j * 16 + l15;
      const int off = colLocal * 32 + (hiHalf ? 16 : 0);
      bHi[j] = *(const v16bf*)&ldsB[buf][0][off];
      bLo[j] = *(const v16bf*)&ldsB[buf][1][off];
    }

#pragma unroll
    for (int i = 0; i < 2; ++i) {
#pragma unroll
      for (int j = 0; j < 2; ++j) {
        acc[i][j] = __builtin_amdgcn_wmma_f32_16x16x32_bf16(
            false, aHi[i], false, bHi[j], (short)0, acc[i][j], false, false);
        acc[i][j] = __builtin_amdgcn_wmma_f32_16x16x32_bf16(
            false, aHi[i], false, bLo[j], (short)0, acc[i][j], false, false);
        acc[i][j] = __builtin_amdgcn_wmma_f32_16x16x32_bf16(
            false, aLo[i], false, bHi[j], (short)0, acc[i][j], false, false);
      }
    }
  }

  // Epilogue: C layout — VGPR v: lanes 0-15 -> M=v, lanes 16-31 -> M=v+8
#pragma unroll
  for (int i = 0; i < 2; ++i) {
    const int r0 = rowBase + i * 16 + (hiHalf ? 8 : 0);
#pragma unroll
    for (int j = 0; j < 2; ++j) {
      const int col = colBase + j * 16 + l15;
      const float bb = bias[col];
#pragma unroll
      for (int e = 0; e < 8; ++e)
        out[(size_t)(r0 + e) * RFF + col] = cosf(acc[i][j][e] + bb) * RFF_SCALE;
    }
  }
}

// ---------------------------------------------------------------------------
// Fallback GEMM (fused f32->bf16 split, no workspace needed) — round-1 path.
// ---------------------------------------------------------------------------
__device__ __forceinline__ void cvt_split16(const float* __restrict__ p0,
                                            const float* __restrict__ p1,
                                            v16bf& hi, v16bf& lo) {
#pragma unroll
  for (int i = 0; i < 8; ++i) {
    float f = p0[i];
    __bf16 h = (__bf16)f;
    hi[i] = h; lo[i] = (__bf16)(f - (float)h);
  }
#pragma unroll
  for (int i = 0; i < 8; ++i) {
    float f = p1[i];
    __bf16 h = (__bf16)f;
    hi[8 + i] = h; lo[8 + i] = (__bf16)(f - (float)h);
  }
}

__global__ __launch_bounds__(256) void rff_gemm_cos_fused(
    const float* __restrict__ x, const float* __restrict__ W,
    const float* __restrict__ bias, float* __restrict__ out)
{
  const int lane = threadIdx.x & 31;
  const int wave = threadIdx.x >> 5;
  const int wm = wave & 3, wn = wave >> 2;
  const int l15 = lane & 15;
  const bool hiHalf = lane >= 16;
  const int rowBase = blockIdx.x * 128 + wm * 32;
  const int colBase = blockIdx.y * 64 + wn * 32;

  v8f acc[2][2];
#pragma unroll
  for (int i = 0; i < 2; ++i)
#pragma unroll
    for (int j = 0; j < 2; ++j)
      acc[i][j] = (v8f){0.f, 0.f, 0.f, 0.f, 0.f, 0.f, 0.f, 0.f};

  const float* xrow[2];
  const float* wrow[2];
#pragma unroll
  for (int i = 0; i < 2; ++i) xrow[i] = x + (size_t)(rowBase + i * 16 + l15) * IN_DIM;
#pragma unroll
  for (int j = 0; j < 2; ++j) wrow[j] = W + (size_t)(colBase + j * 16 + l15) * IN_DIM;

  const int aOff = hiHalf ? 8 : 0;
  const int bOff = hiHalf ? 8 : 0;

  for (int kc = 0; kc < IN_DIM; kc += 32) {
    v16bf aHi[2], aLo[2], bHi[2], bLo[2];
#pragma unroll
    for (int i = 0; i < 2; ++i)
      cvt_split16(xrow[i] + kc + aOff, xrow[i] + kc + aOff + 16, aHi[i], aLo[i]);
#pragma unroll
    for (int j = 0; j < 2; ++j)
      cvt_split16(wrow[j] + kc + bOff, wrow[j] + kc + bOff + 16, bHi[j], bLo[j]);
#pragma unroll
    for (int i = 0; i < 2; ++i) {
#pragma unroll
      for (int j = 0; j < 2; ++j) {
        acc[i][j] = __builtin_amdgcn_wmma_f32_16x16x32_bf16(
            false, aHi[i], false, bHi[j], (short)0, acc[i][j], false, false);
        acc[i][j] = __builtin_amdgcn_wmma_f32_16x16x32_bf16(
            false, aHi[i], false, bLo[j], (short)0, acc[i][j], false, false);
        acc[i][j] = __builtin_amdgcn_wmma_f32_16x16x32_bf16(
            false, aLo[i], false, bHi[j], (short)0, acc[i][j], false, false);
      }
    }
  }

#pragma unroll
  for (int i = 0; i < 2; ++i) {
    const int r0 = rowBase + i * 16 + (hiHalf ? 8 : 0);
#pragma unroll
    for (int j = 0; j < 2; ++j) {
      const int col = colBase + j * 16 + l15;
      const float bb = bias[col];
#pragma unroll
      for (int e = 0; e < 8; ++e)
        out[(size_t)(r0 + e) * RFF + col] = cosf(acc[i][j][e] + bb) * RFF_SCALE;
    }
  }
}

// ---------------------------------------------------------------------------
// Top-K(|z|) mask per row, in place: 4-pass radix select via LDS histogram.
// ---------------------------------------------------------------------------
__global__ __launch_bounds__(256) void topk_mask(float* __restrict__ z)
{
  __shared__ unsigned hist[256];
  __shared__ unsigned sh_prefix;
  __shared__ unsigned sh_want;
  __shared__ unsigned sh_tie;

  const int row = blockIdx.x;
  const int tid = threadIdx.x;
  float* zr = z + (size_t)row * RFF;

  float    vals[16];
  unsigned keys[16];
#pragma unroll
  for (int e = 0; e < 16; ++e) {
    float v = zr[e * 256 + tid];
    vals[e] = v;
    keys[e] = __float_as_uint(fabsf(v));
  }

  if (tid == 0) { sh_prefix = 0u; sh_want = (unsigned)TOPK; }
  __syncthreads();

  unsigned maskHi = 0u;
  for (int shift = 24; shift >= 0; shift -= 8) {
    hist[tid] = 0u;
    __syncthreads();

    const unsigned prefix = sh_prefix;
    const unsigned want   = sh_want;
#pragma unroll
    for (int e = 0; e < 16; ++e) {
      unsigned k = keys[e];
      if ((k & maskHi) == prefix)
        atomicAdd(&hist[(k >> shift) & 255u], 1u);
    }
    __syncthreads();

    if (tid == 0) {
      unsigned cum = 0u;
      int chosen = 0;
      unsigned newWant = want;
      for (int v = 255; v >= 0; --v) {
        unsigned c = hist[v];
        if (cum + c >= want) { chosen = v; newWant = want - cum; break; }
        cum += c;
      }
      sh_prefix = prefix | ((unsigned)chosen << shift);
      sh_want   = newWant;
    }
    __syncthreads();
    maskHi |= (0xFFu << shift);
  }

  const unsigned T    = sh_prefix;
  const unsigned want = sh_want;
  if (tid == 0) sh_tie = 0u;
  __syncthreads();

#pragma unroll
  for (int e = 0; e < 16; ++e) {
    const unsigned k = keys[e];
    float o;
    if (k > T)       o = vals[e];
    else if (k == T) { unsigned ord = atomicAdd(&sh_tie, 1u); o = (ord < want) ? vals[e] : 0.0f; }
    else             o = 0.0f;
    zr[e * 256 + tid] = o;
  }
}

// ---------------------------------------------------------------------------
extern "C" void kernel_launch(void* const* d_in, const int* in_sizes, int n_in,
                              void* d_out, int out_size, void* d_ws, size_t ws_size,
                              hipStream_t stream) {
  (void)in_sizes; (void)n_in; (void)out_size;

  const float* x = (const float*)d_in[0];   // [8192, 512]
  const float* W = (const float*)d_in[1];   // [4096, 512]
  const float* b = (const float*)d_in[2];   // [4096]
  float* out = (float*)d_out;               // [8192, 4096]

  const size_t xElems = (size_t)BATCH * IN_DIM;   // 4,194,304
  const size_t wElems = (size_t)RFF * IN_DIM;     // 2,097,152
  const size_t need   = (xElems + wElems) * 2 * 2; // 24 MB (bf16 hi+lo)

  dim3 gemmGrid(BATCH / 128, RFF / 64);

  if (ws_size >= need) {
    char* ws = (char*)d_ws;
    __bf16* xhp = (__bf16*)ws;
    __bf16* xlp = xhp + xElems;
    __bf16* whp = xlp + xElems;
    __bf16* wlp = whp + wElems;

    split_f32_bf16<<<(int)(xElems / 1024), 256, 0, stream>>>(x, xhp, xlp, (int)xElems);
    split_f32_bf16<<<(int)(wElems / 1024), 256, 0, stream>>>(W, whp, wlp, (int)wElems);
    rff_gemm_cos_bf16<<<gemmGrid, 256, 0, stream>>>(xhp, xlp, whp, wlp, b, out);
  } else {
    rff_gemm_cos_fused<<<gemmGrid, 256, 0, stream>>>(x, W, b, out);
  }

  topk_mask<<<BATCH, 256, 0, stream>>>(out);
}
